// Net_83846351553032
// MI455X (gfx1250) — matrix operand order repaired and checked by reading, not compiled, not run on hardware
//
#include <hip/hip_runtime.h>
#include <hip/hip_bf16.h>
#include <math.h>

#define NNODES 100000
#define NEDGES 1600000
#define FIN    512
#define HEADS  8
#define C1V    8
#define NCLS   40
#define HID    64          // HEADS*C1
#define ETOT   (NEDGES + NNODES)

typedef __attribute__((ext_vector_type(16))) __bf16 v16bf;
typedef __attribute__((ext_vector_type(8)))  float  v8f;

// native RNE convert (v_cvt_*bf16_f32 on gfx1250)
__device__ __forceinline__ __bf16 f2bf(float f) { return (__bf16)f; }
__device__ __forceinline__ float bf2f(__bf16 b) {
  unsigned short u = __builtin_bit_cast(unsigned short, b);
  return __uint_as_float(((unsigned)u) << 16);
}
// order-preserving float<->uint encoding for atomicMax on floats
__device__ __forceinline__ unsigned ford(float f) {
  unsigned u = __float_as_uint(f);
  return (u & 0x80000000u) ? ~u : (u | 0x80000000u);
}
__device__ __forceinline__ float fordinv(unsigned e) {
  return (e & 0x80000000u) ? __uint_as_float(e & 0x7fffffffu)
                           : __uint_as_float(~e);
}

union BV16 { v16bf v; __bf16 e[16]; unsigned short s[16]; };

// ---------------- weight pre-swizzle (bf16 hi/lo split, WMMA B layout) ------
// B 32x16 (16-bit) layout: lane l -> col n = l&15, khalf = l>>4,
// element j (0..15) -> K = (j&8)*2 + khalf*8 + (j&7).
__global__ void prep_w1_k(const float* __restrict__ W, unsigned short* __restrict__ hi,
                          unsigned short* __restrict__ lo) {
  int i = blockIdx.x * blockDim.x + threadIdx.x;     // 16 ktiles * 4 ntiles * 32 * 16
  if (i >= 16 * 4 * 32 * 16) return;
  int j = i & 15, lane = (i >> 4) & 31, nt = (i >> 9) & 3, t = i >> 11;
  int K = t * 32 + ((j & 8) << 1) + (lane >> 4) * 8 + (j & 7);
  int c = nt * 16 + (lane & 15);
  float v = W[(size_t)K * HID + c];
  __bf16 h = f2bf(v);
  hi[i] = __builtin_bit_cast(unsigned short, h);
  lo[i] = __builtin_bit_cast(unsigned short, f2bf(v - bf2f(h)));
}
__global__ void prep_w2_k(const float* __restrict__ W, unsigned short* __restrict__ hi,
                          unsigned short* __restrict__ lo) {
  int i = blockIdx.x * blockDim.x + threadIdx.x;     // 2 ktiles * 3 ntiles * 32 * 16
  if (i >= 2 * 3 * 32 * 16) return;
  int j = i & 15, lane = (i >> 4) & 31, nt = (i >> 9) % 3, t = i / (3 * 512);
  int K = t * 32 + ((j & 8) << 1) + (lane >> 4) * 8 + (j & 7);
  int c = nt * 16 + (lane & 15);
  float v = (c < NCLS) ? W[(size_t)K * NCLS + c] : 0.f;
  __bf16 h = f2bf(v);
  hi[i] = __builtin_bit_cast(unsigned short, h);
  lo[i] = __builtin_bit_cast(unsigned short, f2bf(v - bf2f(h)));
}

// ---------------- GEMM1: h1[N,64] = x[N,512] @ W1 (bf16x3 WMMA) -------------
__global__ void gemm_l1(const float* __restrict__ x,
                        const unsigned short* __restrict__ wph,
                        const unsigned short* __restrict__ wpl,
                        float* __restrict__ h1) {
  int gwave = (int)((blockIdx.x * (unsigned)blockDim.x + threadIdx.x) >> 5);
  int lane  = threadIdx.x & 31;
  if (gwave >= NNODES / 16) return;
  int m0  = gwave << 4;
  int row = m0 + (lane & 15);
  int kh  = lane >> 4;

  union { v8f v; float f[8]; } acc[4];
  #pragma unroll
  for (int n = 0; n < 4; ++n)
    #pragma unroll
    for (int r = 0; r < 8; ++r) acc[n].f[r] = 0.f;

  for (int t = 0; t < 16; ++t) {
    const float* xp = x + (size_t)row * FIN + t * 32 + kh * 8;
    float4 f0 = *(const float4*)(xp);
    float4 f1 = *(const float4*)(xp + 4);
    float4 f2 = *(const float4*)(xp + 16);
    float4 f3 = *(const float4*)(xp + 20);
    float a[16] = {f0.x,f0.y,f0.z,f0.w, f1.x,f1.y,f1.z,f1.w,
                   f2.x,f2.y,f2.z,f2.w, f3.x,f3.y,f3.z,f3.w};
    BV16 ahi, alo;
    #pragma unroll
    for (int j = 0; j < 16; ++j) {
      __bf16 hb = f2bf(a[j]);
      ahi.e[j] = hb;
      alo.e[j] = f2bf(a[j] - bf2f(hb));
    }
    #pragma unroll
    for (int n = 0; n < 4; ++n) {
      int bi = ((t * 4 + n) * 32 + lane) * 16;
      v16bf bh = *(const v16bf*)(wph + bi);
      v16bf bl = *(const v16bf*)(wpl + bi);
      acc[n].v = __builtin_amdgcn_wmma_f32_16x16x32_bf16(false, ahi.v, false, bh, (short)0, acc[n].v, false, false);
      acc[n].v = __builtin_amdgcn_wmma_f32_16x16x32_bf16(false, alo.v, false, bh, (short)0, acc[n].v, false, false);
      acc[n].v = __builtin_amdgcn_wmma_f32_16x16x32_bf16(false, ahi.v, false, bl, (short)0, acc[n].v, false, false);
    }
  }
  int rbase = m0 + kh * 8;
  int col   = lane & 15;
  #pragma unroll
  for (int n = 0; n < 4; ++n)
    #pragma unroll
    for (int r = 0; r < 8; ++r)
      h1[(size_t)(rbase + r) * HID + n * 16 + col] = acc[n].f[r];
}

// ---------------- GEMM2: h2[N,40] = h[N,64] @ W2 (bf16x3 WMMA) --------------
__global__ void gemm_l2(const float* __restrict__ hin,
                        const unsigned short* __restrict__ wph,
                        const unsigned short* __restrict__ wpl,
                        float* __restrict__ h2) {
  int gwave = (int)((blockIdx.x * (unsigned)blockDim.x + threadIdx.x) >> 5);
  int lane  = threadIdx.x & 31;
  if (gwave >= NNODES / 16) return;
  int m0  = gwave << 4;
  int row = m0 + (lane & 15);
  int kh  = lane >> 4;

  union { v8f v; float f[8]; } acc[3];
  #pragma unroll
  for (int n = 0; n < 3; ++n)
    #pragma unroll
    for (int r = 0; r < 8; ++r) acc[n].f[r] = 0.f;

  #pragma unroll
  for (int t = 0; t < 2; ++t) {
    const float* xp = hin + (size_t)row * HID + t * 32 + kh * 8;
    float4 f0 = *(const float4*)(xp);
    float4 f1 = *(const float4*)(xp + 4);
    float4 f2 = *(const float4*)(xp + 16);
    float4 f3 = *(const float4*)(xp + 20);
    float a[16] = {f0.x,f0.y,f0.z,f0.w, f1.x,f1.y,f1.z,f1.w,
                   f2.x,f2.y,f2.z,f2.w, f3.x,f3.y,f3.z,f3.w};
    BV16 ahi, alo;
    #pragma unroll
    for (int j = 0; j < 16; ++j) {
      __bf16 hb = f2bf(a[j]);
      ahi.e[j] = hb;
      alo.e[j] = f2bf(a[j] - bf2f(hb));
    }
    #pragma unroll
    for (int n = 0; n < 3; ++n) {
      int bi = ((t * 3 + n) * 32 + lane) * 16;
      v16bf bh = *(const v16bf*)(wph + bi);
      v16bf bl = *(const v16bf*)(wpl + bi);
      acc[n].v = __builtin_amdgcn_wmma_f32_16x16x32_bf16(false, ahi.v, false, bh, (short)0, acc[n].v, false, false);
      acc[n].v = __builtin_amdgcn_wmma_f32_16x16x32_bf16(false, alo.v, false, bh, (short)0, acc[n].v, false, false);
      acc[n].v = __builtin_amdgcn_wmma_f32_16x16x32_bf16(false, ahi.v, false, bl, (short)0, acc[n].v, false, false);
    }
  }
  int rbase = m0 + kh * 8;
  int col   = lane & 15;
  #pragma unroll
  for (int n = 0; n < 3; ++n)
    #pragma unroll
    for (int r = 0; r < 8; ++r) {
      int c = n * 16 + col;
      if (c < NCLS) h2[(size_t)(rbase + r) * NCLS + c] = acc[n].f[r];
    }
}

// ---------------- attention projections -------------------------------------
__global__ void att_l1(const float* __restrict__ h1, const float* __restrict__ ws,
                       const float* __restrict__ wd, float* __restrict__ as_,
                       float* __restrict__ ad_) {
  int i = blockIdx.x * blockDim.x + threadIdx.x;
  if (i >= NNODES * HEADS) return;
  int node = i >> 3, h = i & 7;
  const float* hp = h1 + (size_t)node * HID + h * C1V;
  float s = 0.f, d = 0.f;
  #pragma unroll
  for (int c = 0; c < C1V; ++c) { float v = hp[c]; s += v * ws[h * C1V + c]; d += v * wd[h * C1V + c]; }
  as_[i] = s; ad_[i] = d;
}
__global__ void att_l2(const float* __restrict__ h2, const float* __restrict__ ws,
                       const float* __restrict__ wd, float* __restrict__ as_,
                       float* __restrict__ ad_) {
  int i = blockIdx.x * blockDim.x + threadIdx.x;
  if (i >= NNODES) return;
  const float* hp = h2 + (size_t)i * NCLS;
  float s = 0.f, d = 0.f;
  #pragma unroll
  for (int c = 0; c < NCLS; ++c) { float v = hp[c]; s += v * ws[c]; d += v * wd[c]; }
  as_[i] = s; ad_[i] = d;
}

// ---------------- edge phase: max / denom / aggregate -----------------------
template <int H>
__global__ void edge_max_k(const long long* __restrict__ ei, const float* __restrict__ as_,
                           const float* __restrict__ ad_, unsigned* __restrict__ m) {
  int e = blockIdx.x * blockDim.x + threadIdx.x;
  if (e >= ETOT) return;
  long long s, d;
  if (e < NEDGES) { s = ei[e]; d = ei[NEDGES + e]; } else { s = d = e - NEDGES; }
  #pragma unroll
  for (int h = 0; h < H; ++h) {
    float v = as_[s * H + h] + ad_[d * H + h];
    v = v > 0.f ? v : 0.2f * v;
    atomicMax(&m[d * H + h], ford(v));
  }
}
template <int H>
__global__ void edge_den_k(const long long* __restrict__ ei, const float* __restrict__ as_,
                           const float* __restrict__ ad_, const unsigned* __restrict__ m,
                           float* __restrict__ den) {
  int e = blockIdx.x * blockDim.x + threadIdx.x;
  if (e >= ETOT) return;
  long long s, d;
  if (e < NEDGES) { s = ei[e]; d = ei[NEDGES + e]; } else { s = d = e - NEDGES; }
  #pragma unroll
  for (int h = 0; h < H; ++h) {
    float v = as_[s * H + h] + ad_[d * H + h];
    v = v > 0.f ? v : 0.2f * v;
    atomicAdd(&den[d * H + h], expf(v - fordinv(m[d * H + h])));
  }
}
template <int H, int C>
__global__ void edge_agg_k(const long long* __restrict__ ei, const float* __restrict__ as_,
                           const float* __restrict__ ad_, const unsigned* __restrict__ m,
                           const float* __restrict__ den, const float* __restrict__ feat,
                           float* __restrict__ out) {
  int e = blockIdx.x * blockDim.x + threadIdx.x;
  if (e >= ETOT) return;
  long long s, d;
  if (e < NEDGES) { s = ei[e]; d = ei[NEDGES + e]; } else { s = d = e - NEDGES; }
  float alpha[H];
  #pragma unroll
  for (int h = 0; h < H; ++h) {
    float v = as_[s * H + h] + ad_[d * H + h];
    v = v > 0.f ? v : 0.2f * v;
    alpha[h] = expf(v - fordinv(m[d * H + h])) / (den[d * H + h] + 1e-16f);
  }
  const float* fp = feat + (size_t)s * (H * C);
  float* op = out + (size_t)d * (H * C);
  #pragma unroll
  for (int i = 0; i < H * C; ++i) atomicAdd(&op[i], fp[i] * alpha[i / C]);
}

// ---------------- pointwise -------------------------------------------------
__global__ void bias_elu_k(float* __restrict__ a, const float* __restrict__ b) {
  int i = blockIdx.x * blockDim.x + threadIdx.x;
  if (i >= NNODES * HID) return;
  float v = a[i] + b[i & (HID - 1)];
  a[i] = v > 0.f ? v : expf(v) - 1.f;
}
__global__ void bias_lsm_k(float* __restrict__ o, const float* __restrict__ b) {
  int i = blockIdx.x * blockDim.x + threadIdx.x;
  if (i >= NNODES) return;
  float* row = o + (size_t)i * NCLS;
  float v[NCLS];
  float mx = -1e30f;
  #pragma unroll
  for (int c = 0; c < NCLS; ++c) { v[c] = row[c] + b[c]; mx = fmaxf(mx, v[c]); }
  float sum = 0.f;
  #pragma unroll
  for (int c = 0; c < NCLS; ++c) sum += expf(v[c] - mx);
  float lse = mx + logf(sum);
  #pragma unroll
  for (int c = 0; c < NCLS; ++c) row[c] = v[c] - lse;
}
__global__ void fill_f_k(float* p, float v, int n) {
  int i = blockIdx.x * blockDim.x + threadIdx.x;
  if (i < n) p[i] = v;
}
__global__ void fill_u_k(unsigned* p, unsigned v, int n) {
  int i = blockIdx.x * blockDim.x + threadIdx.x;
  if (i < n) p[i] = v;
}

static inline int cdiv(long long a, long long b) { return (int)((a + b - 1) / b); }

extern "C" void kernel_launch(void* const* d_in, const int* in_sizes, int n_in,
                              void* d_out, int out_size, void* d_ws, size_t ws_size,
                              hipStream_t stream) {
  (void)in_sizes; (void)n_in; (void)out_size; (void)ws_size;
  const float*     x   = (const float*)d_in[0];
  const long long* ei  = (const long long*)d_in[1];   // int64 edge_index [2,E]
  const float*     W1  = (const float*)d_in[2];
  const float*     as1w= (const float*)d_in[3];
  const float*     ad1w= (const float*)d_in[4];
  const float*     b1  = (const float*)d_in[5];
  const float*     W2  = (const float*)d_in[6];
  const float*     as2w= (const float*)d_in[7];
  const float*     ad2w= (const float*)d_in[8];
  const float*     b2  = (const float*)d_in[9];
  float* out = (float*)d_out;

  char* w = (char*)d_ws;
  auto carve = [&](size_t bytes) { void* p = (void*)w; w += (bytes + 255) & ~(size_t)255; return p; };
  float*          h1   = (float*)carve((size_t)NNODES * HID * 4);
  float*          out1 = (float*)carve((size_t)NNODES * HID * 4);   // agg -> ELU -> h for L2
  float*          h2   = (float*)carve((size_t)NNODES * NCLS * 4);
  float*          as1  = (float*)carve((size_t)NNODES * HEADS * 4);
  float*          ad1  = (float*)carve((size_t)NNODES * HEADS * 4);
  unsigned*       m1   = (unsigned*)carve((size_t)NNODES * HEADS * 4);
  float*          den1 = (float*)carve((size_t)NNODES * HEADS * 4);
  float*          as2  = (float*)carve((size_t)NNODES * 4);
  float*          ad2  = (float*)carve((size_t)NNODES * 4);
  unsigned*       m2   = (unsigned*)carve((size_t)NNODES * 4);
  float*          den2 = (float*)carve((size_t)NNODES * 4);
  unsigned short* w1h  = (unsigned short*)carve(16 * 4 * 32 * 16 * 2);
  unsigned short* w1l  = (unsigned short*)carve(16 * 4 * 32 * 16 * 2);
  unsigned short* w2h  = (unsigned short*)carve(2 * 3 * 32 * 16 * 2);
  unsigned short* w2l  = (unsigned short*)carve(2 * 3 * 32 * 16 * 2);

  const int B = 256;
  const unsigned NEGINF_ENC = 0x007fffffu;            // ford(-inf)

  // weight pre-swizzle
  prep_w1_k<<<cdiv(16 * 4 * 32 * 16, B), B, 0, stream>>>(W1, w1h, w1l);
  prep_w2_k<<<cdiv(2 * 3 * 32 * 16, B), B, 0, stream>>>(W2, w2h, w2l);

  // init accumulators
  fill_u_k<<<cdiv(NNODES * HEADS, B), B, 0, stream>>>(m1, NEGINF_ENC, NNODES * HEADS);
  fill_f_k<<<cdiv(NNODES * HEADS, B), B, 0, stream>>>(den1, 0.f, NNODES * HEADS);
  fill_f_k<<<cdiv(NNODES * HID,   B), B, 0, stream>>>(out1, 0.f, NNODES * HID);
  fill_u_k<<<cdiv(NNODES, B), B, 0, stream>>>(m2, NEGINF_ENC, NNODES);
  fill_f_k<<<cdiv(NNODES, B), B, 0, stream>>>(den2, 0.f, NNODES);
  fill_f_k<<<cdiv(NNODES * NCLS, B), B, 0, stream>>>(out, 0.f, NNODES * NCLS);

  // ---- layer 1 ----
  int waves1 = NNODES / 16;                           // 6250
  gemm_l1<<<cdiv((long long)waves1 * 32, B), B, 0, stream>>>(x, w1h, w1l, h1);
  att_l1<<<cdiv(NNODES * HEADS, B), B, 0, stream>>>(h1, as1w, ad1w, as1, ad1);
  edge_max_k<HEADS><<<cdiv(ETOT, B), B, 0, stream>>>(ei, as1, ad1, m1);
  edge_den_k<HEADS><<<cdiv(ETOT, B), B, 0, stream>>>(ei, as1, ad1, m1, den1);
  edge_agg_k<HEADS, C1V><<<cdiv(ETOT, B), B, 0, stream>>>(ei, as1, ad1, m1, den1, h1, out1);
  bias_elu_k<<<cdiv(NNODES * HID, B), B, 0, stream>>>(out1, b1);

  // ---- layer 2 ----
  gemm_l2<<<cdiv((long long)waves1 * 32, B), B, 0, stream>>>(out1, w2h, w2l, h2);
  att_l2<<<cdiv(NNODES, B), B, 0, stream>>>(h2, as2w, ad2w, as2, ad2);
  edge_max_k<1><<<cdiv(ETOT, B), B, 0, stream>>>(ei, as2, ad2, m2);
  edge_den_k<1><<<cdiv(ETOT, B), B, 0, stream>>>(ei, as2, ad2, m2, den2);
  edge_agg_k<1, NCLS><<<cdiv(ETOT, B), B, 0, stream>>>(ei, as2, ad2, m2, den2, h2, out);
  bias_lsm_k<<<cdiv(NNODES, B), B, 0, stream>>>(out, b2);
}
